// GNNEncoder_36438502539675
// MI455X (gfx1250) — matrix operand order repaired and checked by reading, not compiled, not run on hardware
//
#include <hip/hip_runtime.h>

typedef __attribute__((ext_vector_type(16))) _Float16 v16h;
typedef __attribute__((ext_vector_type(8)))  float    v8f;

#define IN_DIM    256
#define OUT_DIM   64
#define NEG_SLOPE 0.01f
#define LN_EPS    1e-5f

// ---------------------------------------------------------------------------
// Zero the aggregation buffer (d_out) and init degrees to 1.0 (self-loop w).
// ---------------------------------------------------------------------------
__global__ void __launch_bounds__(256) init_kernel(float* __restrict__ deg,
                                                   float* __restrict__ agg, int N) {
    int i = blockIdx.x * blockDim.x + threadIdx.x;
    if (i < N * OUT_DIM) agg[i] = 0.0f;
    if (i < N)           deg[i] = 1.0f;
}

// deg[dst] += ew  (weighted in-degree, self-loops already seeded with 1.0)
__global__ void __launch_bounds__(256) deg_kernel(const int* __restrict__ dst,
                                                  const float* __restrict__ ew,
                                                  float* __restrict__ deg, int E) {
    int i = blockIdx.x * blockDim.x + threadIdx.x;
    if (i < E) atomicAdd(&deg[dst[i]], ew[i]);
}

// deg -> deg^{-1/2} in place
__global__ void __launch_bounds__(256) dinv_kernel(float* __restrict__ deg, int N) {
    int i = blockIdx.x * blockDim.x + threadIdx.x;
    if (i < N) {
        float d = deg[i];
        deg[i] = (d > 0.0f) ? rsqrtf(d) : 0.0f;
    }
}

// ---------------------------------------------------------------------------
// h = x @ W via v_wmma_f32_16x16x32_f16.
// One wave computes a 16x64 row tile: 4 accumulators x 8 K-steps of 32.
// W (32KB as f16) is staged in LDS, pre-swizzled into B-fragment order so a
// lane's v16h fragment is one contiguous 32-byte DS read.
//   A frag (16-bit 16x32): lane<16 row=lane, K = {0..7, 16..23}+k0
//                          lane>=16 row=lane-16, K = {8..15, 24..31}+k0
//   B frag (16-bit 32x16): lanes 0-15 col=lane, K = {0..7, 16..23}+k0
//                          lanes 16-31 col=lane-16, K = {8..15, 24..31}+k0
//   C/D: VGPR r -> row (lane<16 ? r : r+8), col = lane&15
// ---------------------------------------------------------------------------
__global__ void __launch_bounds__(256) gemm_kernel(const float* __restrict__ x,
                                                   const float* __restrict__ W,
                                                   float* __restrict__ h, int N) {
    __shared__ __align__(32) _Float16 Wlds[8 * 4 * 32 * 16]; // [kstep][tile][lane][16]

    // Stage + convert + swizzle W into B-fragment layout.
    for (int idx = threadIdx.x; idx < 16384; idx += 256) {
        int i     = idx & 15;
        int lane  = (idx >> 4) & 31;
        int tile  = (idx >> 9) & 3;
        int kstep = idx >> 11;
        int k = kstep * 32 + ((lane & 16) ? 8 : 0) + (i < 8 ? i : i + 8);
        int n = tile * 16 + (lane & 15);
        Wlds[idx] = (_Float16)W[k * OUT_DIM + n];
    }
    __syncthreads();

    int wid  = (int)((blockIdx.x * blockDim.x + threadIdx.x) >> 5); // global wave = row tile
    int lane = threadIdx.x & 31;
    int row0 = wid * 16;
    if (row0 >= N) return;

    int m   = lane & 15;
    int klo = (lane & 16) ? 8 : 0;                 // A-fragment K sub-offset
    const float* xrow = x + (size_t)(row0 + m) * IN_DIM;

    v8f acc0 = {}, acc1 = {}, acc2 = {}, acc3 = {};

#pragma unroll
    for (int k0 = 0; k0 < IN_DIM; k0 += 32) {
        // A fragment: two contiguous 8-float chunks, convert f32 -> f16.
        float4 f0 = *(const float4*)(xrow + k0 + klo);
        float4 f1 = *(const float4*)(xrow + k0 + klo + 4);
        float4 f2 = *(const float4*)(xrow + k0 + klo + 16);
        float4 f3 = *(const float4*)(xrow + k0 + klo + 20);
        v16h a;
        a[0]  = (_Float16)f0.x; a[1]  = (_Float16)f0.y; a[2]  = (_Float16)f0.z; a[3]  = (_Float16)f0.w;
        a[4]  = (_Float16)f1.x; a[5]  = (_Float16)f1.y; a[6]  = (_Float16)f1.z; a[7]  = (_Float16)f1.w;
        a[8]  = (_Float16)f2.x; a[9]  = (_Float16)f2.y; a[10] = (_Float16)f2.z; a[11] = (_Float16)f2.w;
        a[12] = (_Float16)f3.x; a[13] = (_Float16)f3.y; a[14] = (_Float16)f3.z; a[15] = (_Float16)f3.w;

        const v16h* bp = (const v16h*)&Wlds[(k0 >> 5) * 2048];
        v16h b0 = bp[0 * 32 + lane];
        v16h b1 = bp[1 * 32 + lane];
        v16h b2 = bp[2 * 32 + lane];
        v16h b3 = bp[3 * 32 + lane];

        acc0 = __builtin_amdgcn_wmma_f32_16x16x32_f16(false, a, false, b0, (short)0, acc0, false, false);
        acc1 = __builtin_amdgcn_wmma_f32_16x16x32_f16(false, a, false, b1, (short)0, acc1, false, false);
        acc2 = __builtin_amdgcn_wmma_f32_16x16x32_f16(false, a, false, b2, (short)0, acc2, false, false);
        acc3 = __builtin_amdgcn_wmma_f32_16x16x32_f16(false, a, false, b3, (short)0, acc3, false, false);
    }

    int mrow = row0 + ((lane & 16) ? 8 : 0);
    float* hout = h + (size_t)mrow * OUT_DIM + m;
#pragma unroll
    for (int r = 0; r < 8; ++r) {
        hout[(size_t)r * OUT_DIM + 0]  = acc0[r];
        hout[(size_t)r * OUT_DIM + 16] = acc1[r];
        hout[(size_t)r * OUT_DIM + 32] = acc2[r];
        hout[(size_t)r * OUT_DIM + 48] = acc3[r];
    }
}

// ---------------------------------------------------------------------------
// Scatter: one wave per (edge or self-loop); each lane handles 2 of 64 dims.
// h (25.6MB) and agg (25.6MB) are L2-resident on MI455X (192MB L2), so this
// runs at L2 gather / atomic throughput rather than HBM.
// ---------------------------------------------------------------------------
__global__ void __launch_bounds__(256) scatter_kernel(const int* __restrict__ src,
                                                      const int* __restrict__ dst,
                                                      const float* __restrict__ ew,
                                                      const float* __restrict__ dinv,
                                                      const float* __restrict__ h,
                                                      float* __restrict__ agg,
                                                      int E, int N) {
    int wid  = (int)((blockIdx.x * blockDim.x + threadIdx.x) >> 5);
    int lane = threadIdx.x & 31;
    if (wid >= E + N) return;

    int s, d;
    float nrm;
    if (wid < E) {
        s = src[wid];
        d = dst[wid];
        nrm = dinv[s] * ew[wid] * dinv[d];
    } else {                       // self-loop with weight 1
        s = wid - E;
        d = s;
        float di = dinv[s];
        nrm = di * di;
    }

    float2 v = *(const float2*)(h + (size_t)s * OUT_DIM + lane * 2);
    float* ap = agg + (size_t)d * OUT_DIM + lane * 2;
    atomicAdd(ap,     v.x * nrm);
    atomicAdd(ap + 1, v.y * nrm);
}

// ---------------------------------------------------------------------------
// Epilogue: bias + leaky-ReLU + LayerNorm, one wave per node, 2 dims/lane,
// wave32 shuffle reduction. In-place on d_out.
// ---------------------------------------------------------------------------
__global__ void __launch_bounds__(256) post_kernel(const float* __restrict__ bias,
                                                   const float* __restrict__ gamma,
                                                   const float* __restrict__ beta,
                                                   float* __restrict__ out, int N) {
    int node = (int)((blockIdx.x * blockDim.x + threadIdx.x) >> 5);
    int lane = threadIdx.x & 31;
    if (node >= N) return;

    float* row = out + (size_t)node * OUT_DIM;
    float2 v  = *(const float2*)(row + lane * 2);
    float2 bb = *(const float2*)(bias + lane * 2);
    float y0 = v.x + bb.x;
    float y1 = v.y + bb.y;
    y0 = (y0 >= 0.0f) ? y0 : NEG_SLOPE * y0;
    y1 = (y1 >= 0.0f) ? y1 : NEG_SLOPE * y1;

    float s  = y0 + y1;
    float ss = y0 * y0 + y1 * y1;
#pragma unroll
    for (int off = 16; off > 0; off >>= 1) {
        s  += __shfl_xor(s,  off, 32);
        ss += __shfl_xor(ss, off, 32);
    }
    float mu   = s * (1.0f / OUT_DIM);
    float var  = ss * (1.0f / OUT_DIM) - mu * mu;
    float rinv = rsqrtf(var + LN_EPS);

    float2 g  = *(const float2*)(gamma + lane * 2);
    float2 bt = *(const float2*)(beta  + lane * 2);
    float2 o;
    o.x = (y0 - mu) * rinv * g.x + bt.x;
    o.y = (y1 - mu) * rinv * g.y + bt.y;
    *(float2*)(row + lane * 2) = o;
}

// ---------------------------------------------------------------------------
extern "C" void kernel_launch(void* const* d_in, const int* in_sizes, int n_in,
                              void* d_out, int out_size, void* d_ws, size_t ws_size,
                              hipStream_t stream) {
    const float* x     = (const float*)d_in[0];
    const int*   eidx  = (const int*)  d_in[1];
    const float* ew    = (const float*)d_in[2];
    const float* W     = (const float*)d_in[3];
    const float* bias  = (const float*)d_in[4];
    const float* gamma = (const float*)d_in[5];
    const float* beta  = (const float*)d_in[6];

    const int N = in_sizes[0] / IN_DIM;
    const int E = in_sizes[2];
    const int* src = eidx;
    const int* dst = eidx + E;

    float* agg = (float*)d_out;

    // Workspace: deg/dinv [N] then h [N*64]
    float* deg = (float*)d_ws;
    size_t degBytes = (((size_t)N * sizeof(float)) + 255) & ~(size_t)255;
    float* h = (float*)((char*)d_ws + degBytes);

    const int T = 256;
    init_kernel<<<(N * OUT_DIM + T - 1) / T, T, 0, stream>>>(deg, agg, N);
    deg_kernel <<<(E + T - 1) / T, T, 0, stream>>>(dst, ew, deg, E);
    dinv_kernel<<<(N + T - 1) / T, T, 0, stream>>>(deg, N);

    int rowTiles   = (N + 15) / 16;          // one wave per 16-row tile
    int gemmBlocks = (rowTiles + 7) / 8;     // 8 waves per 256-thread block
    gemm_kernel<<<gemmBlocks, T, 0, stream>>>(x, W, h, N);

    long long waves = (long long)E + N;      // edges + self-loops
    int scatterBlocks = (int)((waves + 7) / 8);
    scatter_kernel<<<scatterBlocks, T, 0, stream>>>(src, dst, ew, deg, h, agg, E, N);

    post_kernel<<<(N + 7) / 8, T, 0, stream>>>(bias, gamma, beta, agg, N);
}
